// TransformerBlock_21397527069202
// MI455X (gfx1250) — matrix operand order repaired and checked by reading, not compiled, not run on hardware
//
#include <hip/hip_runtime.h>
#include <hip/hip_bf16.h>
#include <stdint.h>

// ---------------- problem constants ----------------
#define T_TOK 1024
#define H_DIM 2880
#define NHEAD 64
#define NKVH  8
#define HDIM  64
#define NEXP  8
#define DINT  2880
#define WINDOW 128
#define ALPHA_C 1.702f
#define LIMIT_C 7.0f

typedef __attribute__((ext_vector_type(16))) __bf16 v16bf;
typedef __attribute__((ext_vector_type(8)))  __bf16 v8bf;
typedef __attribute__((ext_vector_type(8)))  float  v8f;

// ---------------- WMMA tile config ----------------
#define BM 128
#define BN 128
#define BK 32
#define STR 40     // LDS inner stride (bf16 elems): 80B rows, 16B-aligned chunks

__device__ __forceinline__ v8bf lds_ld8(const __bf16* p) { return *(const v8bf*)p; }
__device__ __forceinline__ void lds_st8(__bf16* p, v8bf v) { *(v8bf*)p = v; }
__device__ __forceinline__ v16bf cat16(v8bf lo, v8bf hi)
{
    return __builtin_shufflevector(lo, hi, 0,1,2,3,4,5,6,7,8,9,10,11,12,13,14,15);
}

// gfx1250 async global->LDS copy (ASYNCcnt-tracked). VDST = LDS byte
// address (low 32 bits of the generic LDS pointer), then 64-bit global VA.
__device__ __forceinline__ void async_g2l_b128(const void* gptr, void* lptr)
{
    uint32_t loff = (uint32_t)(uintptr_t)lptr;
    asm volatile("global_load_async_to_lds_b128 %0, %1, off"
                 :: "v"(loff), "v"(gptr) : "memory");
}
__device__ __forceinline__ void wait_async0()
{
    asm volatile("s_wait_asynccnt 0x0" ::: "memory");
}

// ===================================================================
// RMSNorm: one block per token row; writes bf16 (WMMA A operand) and
// optionally fp32 (router input).
// ===================================================================
__global__ __launch_bounds__(256)
void rmsnorm_k(const float* __restrict__ x, const float* __restrict__ w,
               __bf16* __restrict__ obf, float* __restrict__ of32)
{
    __shared__ float red[256];
    const int t = blockIdx.x, tid = threadIdx.x;
    const float* row = x + (size_t)t * H_DIM;
    float s = 0.f;
    for (int i = tid; i < H_DIM; i += 256) { float v = row[i]; s += v * v; }
    red[tid] = s; __syncthreads();
    for (int st = 128; st > 0; st >>= 1) {
        if (tid < st) red[tid] += red[tid + st];
        __syncthreads();
    }
    const float r = rsqrtf(red[0] / (float)H_DIM + 1e-5f);
    for (int i = tid; i < H_DIM; i += 256) {
        float v = row[i] * r * w[i];
        obf[(size_t)t * H_DIM + i] = (__bf16)v;
        if (of32) of32[(size_t)t * H_DIM + i] = v;
    }
}

// ===================================================================
// Generic WMMA GEMM:  out[M,N] = A_bf16[M,K] * W_f32[K,N] (+bias)(+resid)
// Ping-pong LDS tiles; A tile staged via global_load_async_to_lds_b128
// (bf16, already in final layout), B tile via register pipeline with
// packed bf16 conversion. 8 v_wmma_f32_16x16x32_bf16 per K-step.
// ===================================================================
__global__ __launch_bounds__(256)
void gemm_bf16_f32w(const __bf16* __restrict__ A, const float* __restrict__ W,
                    const float* __restrict__ bias, const float* __restrict__ resid,
                    float* __restrict__ out, int M, int N, int K)
{
    __shared__ __attribute__((aligned(16))) __bf16 As[2][BM * STR];
    __shared__ __attribute__((aligned(16))) __bf16 Bs[2][BN * STR];
    const int tid  = threadIdx.x;
    const int lane = tid & 31, wv = tid >> 5;
    const int n0 = blockIdx.x * BN, m0 = blockIdx.y * BM;

    v8f acc[8];
    #pragma unroll
    for (int i = 0; i < 8; i++)
        #pragma unroll
        for (int j = 0; j < 8; j++) acc[i][j] = 0.f;

    const int am   = lane & 15;          // A-frag row within wave tile
    const int koff = (lane >> 4) * 8;    // unused; A-frag K base is kb
    const int bn   = lane & 15;          // B-frag column
    const int kb   = (lane >> 4) * 16;   // B-frag K base per half-wave

    // staging roles: 256 threads = (index 0..127) x (k-half 0..1)
    const int srow  = tid >> 1;
    const int shalf = tid & 1;
    const int scol  = n0 + srow;
    const bool colok = scol < N;
    const int sdst = srow * STR + shalf * 16;

    const int ksteps = K / BK;

    // ---- prologue: stage tile 0 into buffer 0 ----
    {
        const __bf16* ap = &A[(size_t)(m0 + srow) * K + shalf * 16];
        async_g2l_b128(ap,     &As[0][sdst]);
        async_g2l_b128(ap + 8, &As[0][sdst + 8]);
        v8bf b0, b1;
        if (colok) {
            const float* wp = &W[(size_t)(shalf * 16) * N + scol];
            #pragma unroll
            for (int i = 0; i < 8; i++) b0[i] = (__bf16)wp[(size_t)i * N];
            #pragma unroll
            for (int i = 0; i < 8; i++) b1[i] = (__bf16)wp[(size_t)(i + 8) * N];
        } else {
            #pragma unroll
            for (int i = 0; i < 8; i++) { b0[i] = (__bf16)0.f; b1[i] = (__bf16)0.f; }
        }
        lds_st8(&Bs[0][sdst], b0);
        lds_st8(&Bs[0][sdst + 8], b1);
        wait_async0();
    }
    __syncthreads();

    for (int kt = 0; kt < ksteps; kt++) {
        const int cur = kt & 1;
        const bool more = (kt + 1) < ksteps;

        // ---- phase 1: next tile -- async A copy + B global loads ----
        float nw[16];
        if (more) {
            const int k1 = (kt + 1) * BK;
            const __bf16* ap = &A[(size_t)(m0 + srow) * K + k1 + shalf * 16];
            async_g2l_b128(ap,     &As[cur ^ 1][sdst]);
            async_g2l_b128(ap + 8, &As[cur ^ 1][sdst + 8]);
            if (colok) {
                const float* wp = &W[(size_t)(k1 + shalf * 16) * N + scol];
                #pragma unroll
                for (int i = 0; i < 16; i++) nw[i] = wp[(size_t)i * N];
            }
            __builtin_prefetch(&A[(size_t)(m0 + srow) * K + k1 + BK], 0, 1);
        }

        // ---- phase 2: compute current tile ----
        const __bf16* arow = &As[cur][(wv * 16 + am) * STR];
        v16bf afrag = cat16(lds_ld8(arow + kb), lds_ld8(arow + kb + 8));
        v16bf bfrag[8];
        #pragma unroll
        for (int nt = 0; nt < 8; nt++) {
            const __bf16* brow = &Bs[cur][(nt * 16 + bn) * STR + kb];
            bfrag[nt] = cat16(lds_ld8(brow), lds_ld8(brow + 8));
        }
        #pragma unroll
        for (int nt = 0; nt < 8; nt++) {
            acc[nt] = __builtin_amdgcn_wmma_f32_16x16x32_bf16(
                false, afrag, false, bfrag[nt], (short)0, acc[nt], false, false);
        }

        // ---- phase 3: convert + store next B tile ----
        if (more) {
            const int nxt = cur ^ 1;
            v8bf b0, b1;
            if (colok) {
                #pragma unroll
                for (int i = 0; i < 8; i++) b0[i] = (__bf16)nw[i];
                #pragma unroll
                for (int i = 0; i < 8; i++) b1[i] = (__bf16)nw[i + 8];
            } else {
                #pragma unroll
                for (int i = 0; i < 8; i++) { b0[i] = (__bf16)0.f; b1[i] = (__bf16)0.f; }
            }
            lds_st8(&Bs[nxt][sdst], b0);
            lds_st8(&Bs[nxt][sdst + 8], b1);
        }
        wait_async0();
        __syncthreads();
    }

    // ---- epilogue: C/D layout (lane: N=lane&15, M=j+8*(lane>>4)) ----
    const int cn = lane & 15;
    const int mh = (lane >> 4) * 8;
    #pragma unroll
    for (int nt = 0; nt < 8; nt++) {
        int n = n0 + nt * 16 + cn;
        if (n < N) {
            float b = bias ? bias[n] : 0.f;
            #pragma unroll
            for (int j = 0; j < 8; j++) {
                int m = m0 + wv * 16 + mh + j;
                float v = acc[nt][j] + b;
                if (resid) v += resid[(size_t)m * N + n];
                out[(size_t)m * N + n] = v;
            }
        }
    }
}

// ===================================================================
// MXFP4 WMMA GEMM: B is E2M1 nibbles with E8M0 block scales, blocks of
// 32 along K (one block per K-step per column). Raw bytes fetched in
// phase 1, dequant deferred to phase 3 (overlaps WMMA). A tile staged
// with async global->LDS copies.
// Storage: blocks[n][K/32][16] bytes, scales[n][K/32].
// mode: rs==nullptr -> out = acc + bias
//       rs!=nullptr -> out += rs[m*8+e] * (acc + bias)
// ===================================================================
__device__ __forceinline__ float e2m1_val(int nib)
{
    int mm = nib & 1, ex = (nib >> 1) & 3;
    float mag = (ex == 0) ? 0.5f * (float)mm
                          : (1.0f + 0.5f * (float)mm) * (float)(1 << (ex - 1));
    return (nib & 8) ? -mag : mag;
}

__device__ __forceinline__ void dequant8(uint32_t w, float scalef, v8bf* o)
{
    #pragma unroll
    for (int b = 0; b < 4; b++) {
        uint32_t by = (w >> (8 * b)) & 0xff;
        (*o)[2 * b]     = (__bf16)(e2m1_val(by & 15) * scalef);
        (*o)[2 * b + 1] = (__bf16)(e2m1_val(by >> 4) * scalef);
    }
}

__global__ __launch_bounds__(256)
void gemm_bf16_mxfp4(const __bf16* __restrict__ A,
                     const uint8_t* __restrict__ blocks,
                     const uint8_t* __restrict__ scales,
                     const float* __restrict__ bias,
                     const float* __restrict__ rs, int eidx,
                     float* __restrict__ out, int M, int N, int K)
{
    __shared__ __attribute__((aligned(16))) __bf16 As[2][BM * STR];
    __shared__ __attribute__((aligned(16))) __bf16 Bs[2][BN * STR];
    const int tid  = threadIdx.x;
    const int lane = tid & 31, wv = tid >> 5;
    const int n0 = blockIdx.x * BN, m0 = blockIdx.y * BM;
    const int kblocks = K >> 5;

    v8f acc[8];
    #pragma unroll
    for (int i = 0; i < 8; i++)
        #pragma unroll
        for (int j = 0; j < 8; j++) acc[i][j] = 0.f;

    const int am   = lane & 15;
    const int koff = (lane >> 4) * 8;
    const int bn   = lane & 15;
    const int kb   = (lane >> 4) * 16;

    const int srow  = tid >> 1;
    const int shalf = tid & 1;
    const int scol  = n0 + srow;
    const bool colok = scol < N;
    const int sdst = srow * STR + shalf * 16;
    const size_t bbase = (size_t)(colok ? scol : 0) * kblocks;

    const int ksteps = K / BK;

    // ---- prologue: stage tile 0 ----
    {
        const __bf16* ap = &A[(size_t)(m0 + srow) * K + shalf * 16];
        async_g2l_b128(ap,     &As[0][sdst]);
        async_g2l_b128(ap + 8, &As[0][sdst + 8]);
        v8bf b0, b1;
        if (colok) {
            const uint32_t* bp = (const uint32_t*)(blocks + bbase * 16) + shalf * 2;
            float scalef = __uint_as_float((uint32_t)scales[bbase] << 23);
            dequant8(bp[0], scalef, &b0);
            dequant8(bp[1], scalef, &b1);
        } else {
            #pragma unroll
            for (int i = 0; i < 8; i++) { b0[i] = (__bf16)0.f; b1[i] = (__bf16)0.f; }
        }
        lds_st8(&Bs[0][sdst], b0);
        lds_st8(&Bs[0][sdst + 8], b1);
        wait_async0();
    }
    __syncthreads();

    for (int kt = 0; kt < ksteps; kt++) {
        const int cur = kt & 1;
        const bool more = (kt + 1) < ksteps;

        // ---- phase 1: next tile -- async A copy + raw B block fetch ----
        uint32_t nb0 = 0, nb1 = 0;
        float nscale = 0.f;
        if (more) {
            const int k1 = (kt + 1) * BK;
            const __bf16* ap = &A[(size_t)(m0 + srow) * K + k1 + shalf * 16];
            async_g2l_b128(ap,     &As[cur ^ 1][sdst]);
            async_g2l_b128(ap + 8, &As[cur ^ 1][sdst + 8]);
            if (colok) {
                size_t boff = bbase + (k1 >> 5);
                const uint32_t* bp = (const uint32_t*)(blocks + boff * 16) + shalf * 2;
                nb0 = bp[0]; nb1 = bp[1];
                nscale = __uint_as_float((uint32_t)scales[boff] << 23);
            }
            __builtin_prefetch(&A[(size_t)(m0 + srow) * K + k1 + BK], 0, 1);
        }

        // ---- phase 2: compute current tile ----
        const __bf16* arow = &As[cur][(wv * 16 + am) * STR];
        v16bf afrag = cat16(lds_ld8(arow + kb), lds_ld8(arow + kb + 8));
        v16bf bfrag[8];
        #pragma unroll
        for (int nt = 0; nt < 8; nt++) {
            const __bf16* brow = &Bs[cur][(nt * 16 + bn) * STR + kb];
            bfrag[nt] = cat16(lds_ld8(brow), lds_ld8(brow + 8));
        }
        #pragma unroll
        for (int nt = 0; nt < 8; nt++) {
            acc[nt] = __builtin_amdgcn_wmma_f32_16x16x32_bf16(
                false, afrag, false, bfrag[nt], (short)0, acc[nt], false, false);
        }

        // ---- phase 3: dequant + store next B tile ----
        if (more) {
            const int nxt = cur ^ 1;
            v8bf b0, b1;
            if (colok) {
                dequant8(nb0, nscale, &b0);
                dequant8(nb1, nscale, &b1);
            } else {
                #pragma unroll
                for (int i = 0; i < 8; i++) { b0[i] = (__bf16)0.f; b1[i] = (__bf16)0.f; }
            }
            lds_st8(&Bs[nxt][sdst], b0);
            lds_st8(&Bs[nxt][sdst + 8], b1);
        }
        wait_async0();
        __syncthreads();
    }

    const int cn = lane & 15;
    const int mh = (lane >> 4) * 8;
    #pragma unroll
    for (int nt = 0; nt < 8; nt++) {
        int n = n0 + nt * 16 + cn;
        if (n < N) {
            float b = bias ? bias[n] : 0.f;
            #pragma unroll
            for (int j = 0; j < 8; j++) {
                int m = m0 + wv * 16 + mh + j;
                float v = acc[nt][j] + b;
                if (rs) {
                    out[(size_t)m * N + n] += rs[(size_t)m * NEXP + eidx] * v;
                } else {
                    out[(size_t)m * N + n] = v;
                }
            }
        }
    }
}

// ===================================================================
// YaRN RoPE, in place on q or k laid out [T, nheads, 64].
// ===================================================================
__global__ void rope_k(float* __restrict__ x, const int* __restrict__ positions,
                       int nheads, int total)
{
    int idx = blockIdx.x * blockDim.x + threadIdx.x;
    if (idx >= total) return;
    int d = idx & 31;
    int h = (idx >> 5) % nheads;
    int t = idx / (32 * nheads);

    const float lnbase = logf(150000.0f);
    float freq = __expf(lnbase * (2.0f * (float)d) / 64.0f);
    float conc = 0.1f * logf(32.0f) + 1.0f;
    float low  = 32.0f * logf(4096.0f / (32.0f * 6.2831853f)) / lnbase;
    float high = 32.0f * logf(4096.0f / (1.0f * 6.2831853f)) / lnbase;
    float ramp = ((float)d - low) / (high - low);
    ramp = fminf(fmaxf(ramp, 0.f), 1.f);
    float mask = 1.0f - ramp;
    float invf = (1.0f - mask) / (32.0f * freq) + mask / freq;
    float ang = (float)positions[t] * invf;
    float c = cosf(ang) * conc, s = sinf(ang) * conc;

    size_t base = ((size_t)t * nheads + h) * 64;
    float x1 = x[base + d], x2 = x[base + 32 + d];
    x[base + d]      = x1 * c - x2 * s;
    x[base + 32 + d] = x2 * c + x1 * s;
}

// ===================================================================
// Sliding-window causal attention with sink, one wave per (query, head).
// ===================================================================
__global__ __launch_bounds__(256)
void attn_k(const float* __restrict__ q, const float* __restrict__ k,
            const float* __restrict__ v, const float* __restrict__ sinks,
            __bf16* __restrict__ out)
{
    __shared__ float probs[8][WINDOW];
    const int lane = threadIdx.x & 31, wv = threadIdx.x >> 5;
    const int pair = blockIdx.x * 8 + wv;
    const int h  = pair & (NHEAD - 1);
    const int qi = pair >> 6;
    const int kvh = h >> 3;                 // GQA repeat = 8
    const int jstart = (qi >= WINDOW - 1) ? qi - (WINDOW - 1) : 0;
    const int nk = qi - jstart + 1;

    const float* qp = q + ((size_t)qi * NHEAD + h) * HDIM;
    float sc[4];
    #pragma unroll
    for (int c = 0; c < 4; c++) {
        int jj = c * 32 + lane;
        sc[c] = -1e30f;
        if (jj < nk) {
            const float* kp = k + ((size_t)(jstart + jj) * NKVH + kvh) * HDIM;
            float d = 0.f;
            #pragma unroll
            for (int i = 0; i < 16; i++) {
                float4 qa = ((const float4*)qp)[i];
                float4 ka = ((const float4*)kp)[i];
                d += qa.x * ka.x + qa.y * ka.y + qa.z * ka.z + qa.w * ka.w;
            }
            sc[c] = d * 0.125f;   // 1/sqrt(64)
        }
    }
    const float snk = sinks[h];
    float m = fmaxf(fmaxf(sc[0], sc[1]), fmaxf(sc[2], sc[3]));
    #pragma unroll
    for (int off = 16; off > 0; off >>= 1) m = fmaxf(m, __shfl_xor(m, off, 32));
    m = fmaxf(m, snk);
    float psum = 0.f;
    #pragma unroll
    for (int c = 0; c < 4; c++) {
        float p = (sc[c] > -1e29f) ? __expf(sc[c] - m) : 0.f;
        sc[c] = p; psum += p;
    }
    #pragma unroll
    for (int off = 16; off > 0; off >>= 1) psum += __shfl_xor(psum, off, 32);
    const float inv = 1.0f / (psum + __expf(snk - m));
    #pragma unroll
    for (int c = 0; c < 4; c++) probs[wv][c * 32 + lane] = sc[c] * inv;
    __syncthreads();

    const int d0 = lane * 2;
    float a0 = 0.f, a1 = 0.f;
    for (int jj = 0; jj < nk; jj++) {
        float pr = probs[wv][jj];
        const float* vp = v + ((size_t)(jstart + jj) * NKVH + kvh) * HDIM;
        a0 += pr * vp[d0];
        a1 += pr * vp[d0 + 1];
    }
    size_t ob = ((size_t)qi * NHEAD + h) * HDIM;
    out[ob + d0]     = (__bf16)a0;
    out[ob + d0 + 1] = (__bf16)a1;
}

// ===================================================================
// Router: logits = t2 @ router_w + b ; top-4 softmax scatter.
// ===================================================================
__global__ void router_k(const float* __restrict__ t2, const float* __restrict__ rw,
                         const float* __restrict__ rb, float* __restrict__ rs)
{
    int t = blockIdx.x * blockDim.x + threadIdx.x;
    if (t >= T_TOK) return;
    float logit[8];
    #pragma unroll
    for (int e = 0; e < 8; e++) logit[e] = rb[e];
    const float* row = t2 + (size_t)t * H_DIM;
    for (int i = 0; i < H_DIM; i++) {
        float x = row[i];
        #pragma unroll
        for (int e = 0; e < 8; e++) logit[e] += x * rw[i * 8 + e];
    }
    float tv[4]; int ti[4];
    #pragma unroll
    for (int kk = 0; kk < 4; kk++) {
        float best = -1e30f; int bi = 0;
        #pragma unroll
        for (int e = 0; e < 8; e++)
            if (logit[e] > best) { best = logit[e]; bi = e; }
        tv[kk] = best; ti[kk] = bi; logit[bi] = -1e38f;
    }
    float se = 0.f, pe[4];
    #pragma unroll
    for (int kk = 0; kk < 4; kk++) { pe[kk] = __expf(tv[kk] - tv[0]); se += pe[kk]; }
    float o[8];
    #pragma unroll
    for (int e = 0; e < 8; e++) o[e] = 0.f;
    #pragma unroll
    for (int kk = 0; kk < 4; kk++) o[ti[kk]] = pe[kk] / se;
    #pragma unroll
    for (int e = 0; e < 8; e++) rs[(size_t)t * 8 + e] = o[e];
}

// ===================================================================
// SwiGLU with clamps; writes bf16 A operand for the down GEMM.
// ===================================================================
__global__ void glu_k(const float* __restrict__ gu, __bf16* __restrict__ act, int total)
{
    int idx = blockIdx.x * blockDim.x + threadIdx.x;
    if (idx >= total) return;
    int t = idx / DINT, d = idx - t * DINT;
    float g = gu[(size_t)t * (2 * DINT) + 2 * d];
    float u = gu[(size_t)t * (2 * DINT) + 2 * d + 1];
    g = fminf(g, LIMIT_C);
    u = fminf(fmaxf(u, -LIMIT_C), LIMIT_C);
    float glu = g / (1.0f + __expf(-g * ALPHA_C));
    act[idx] = (__bf16)((u + 1.0f) * glu);
}

// ===================================================================
// host-side orchestration
// ===================================================================
static inline size_t alignup(size_t x) { return (x + 255) & ~(size_t)255; }

extern "C" void kernel_launch(void* const* d_in, const int* in_sizes, int n_in,
                              void* d_out, int out_size, void* d_ws, size_t ws_size,
                              hipStream_t stream)
{
    const float*   hidden    = (const float*)d_in[0];
    const int*     positions = (const int*)d_in[1];
    const float*   norm1_w   = (const float*)d_in[2];
    const float*   q_w       = (const float*)d_in[3];
    const float*   q_b       = (const float*)d_in[4];
    const float*   k_w       = (const float*)d_in[5];
    const float*   k_b       = (const float*)d_in[6];
    const float*   v_w       = (const float*)d_in[7];
    const float*   v_b       = (const float*)d_in[8];
    const float*   sinks     = (const float*)d_in[9];
    const float*   o_w       = (const float*)d_in[10];
    const float*   o_b       = (const float*)d_in[11];
    const float*   norm2_w   = (const float*)d_in[12];
    const float*   router_w  = (const float*)d_in[13];
    const float*   router_b  = (const float*)d_in[14];
    const float*   gu_bias   = (const float*)d_in[15];
    const float*   dn_bias   = (const float*)d_in[16];
    const uint8_t* gu_scales = (const uint8_t*)d_in[17];
    const uint8_t* gu_blocks = (const uint8_t*)d_in[18];
    const uint8_t* dn_scales = (const uint8_t*)d_in[19];
    const uint8_t* dn_blocks = (const uint8_t*)d_in[20];
    float* out = (float*)d_out;

    // -------- workspace carve-up --------
    char* ws = (char*)d_ws;
    size_t off = 0;
    __bf16* t1bf   = (__bf16*)(ws + off); off = alignup(off + (size_t)T_TOK * H_DIM * 2);
    __bf16* t2bf   = (__bf16*)(ws + off); off = alignup(off + (size_t)T_TOK * H_DIM * 2);
    float*  t2f    = (float*) (ws + off); off = alignup(off + (size_t)T_TOK * H_DIM * 4);
    float*  qbuf   = (float*) (ws + off); off = alignup(off + (size_t)T_TOK * NHEAD * HDIM * 4);
    float*  kbuf   = (float*) (ws + off); off = alignup(off + (size_t)T_TOK * NKVH * HDIM * 4);
    float*  vbuf   = (float*) (ws + off); off = alignup(off + (size_t)T_TOK * NKVH * HDIM * 4);
    __bf16* attnbf = (__bf16*)(ws + off); off = alignup(off + (size_t)T_TOK * NHEAD * HDIM * 2);
    float*  rsbuf  = (float*) (ws + off); off = alignup(off + (size_t)T_TOK * NEXP * 4);
    float*  gubuf  = (float*) (ws + off); off = alignup(off + (size_t)T_TOK * 2 * DINT * 4);
    __bf16* actbf  = (__bf16*)(ws + off); off = alignup(off + (size_t)T_TOK * DINT * 2);
    (void)ws_size; (void)n_in; (void)in_sizes; (void)out_size;

    // 1) rmsnorm1 -> t1 (bf16)
    rmsnorm_k<<<T_TOK, 256, 0, stream>>>(hidden, norm1_w, t1bf, nullptr);

    // 2) QKV projections (WMMA bf16)
    gemm_bf16_f32w<<<dim3(4096 / BN, T_TOK / BM), 256, 0, stream>>>(
        t1bf, q_w, q_b, nullptr, qbuf, T_TOK, 4096, H_DIM);
    gemm_bf16_f32w<<<dim3((512 + BN - 1) / BN, T_TOK / BM), 256, 0, stream>>>(
        t1bf, k_w, k_b, nullptr, kbuf, T_TOK, 512, H_DIM);
    gemm_bf16_f32w<<<dim3((512 + BN - 1) / BN, T_TOK / BM), 256, 0, stream>>>(
        t1bf, v_w, v_b, nullptr, vbuf, T_TOK, 512, H_DIM);

    // 3) YaRN RoPE on q and k
    {
        int totq = T_TOK * NHEAD * 32;
        rope_k<<<(totq + 255) / 256, 256, 0, stream>>>(qbuf, positions, NHEAD, totq);
        int totk = T_TOK * NKVH * 32;
        rope_k<<<(totk + 255) / 256, 256, 0, stream>>>(kbuf, positions, NKVH, totk);
    }

    // 4) windowed attention with sinks -> attn (bf16)
    attn_k<<<(T_TOK * NHEAD) / 8, 256, 0, stream>>>(qbuf, kbuf, vbuf, sinks, attnbf);

    // 5) O projection + residual -> d_out holds h
    gemm_bf16_f32w<<<dim3((H_DIM + BN - 1) / BN, T_TOK / BM), 256, 0, stream>>>(
        attnbf, o_w, o_b, hidden, out, T_TOK, H_DIM, 4096);

    // 6) rmsnorm2 -> t2 (bf16 + f32)
    rmsnorm_k<<<T_TOK, 256, 0, stream>>>(out, norm2_w, t2bf, t2f);

    // 7) router top-4 softmax scores
    router_k<<<(T_TOK + 255) / 256, 256, 0, stream>>>(t2f, router_w, router_b, rsbuf);

    // 8) MoE: all experts (as in reference); sequential launches accumulate
    //    score*(down(glu(gate_up(t2)))+bias) into d_out race-free.
    const size_t guBlkStride = (size_t)(2 * DINT) * (H_DIM / 32) * 16;
    const size_t guSclStride = (size_t)(2 * DINT) * (H_DIM / 32);
    const size_t dnBlkStride = (size_t)H_DIM * (DINT / 32) * 16;
    const size_t dnSclStride = (size_t)H_DIM * (DINT / 32);
    for (int e = 0; e < NEXP; e++) {
        gemm_bf16_mxfp4<<<dim3((2 * DINT + BN - 1) / BN, T_TOK / BM), 256, 0, stream>>>(
            t2bf, gu_blocks + e * guBlkStride, gu_scales + e * guSclStride,
            gu_bias + (size_t)e * 2 * DINT, nullptr, e,
            gubuf, T_TOK, 2 * DINT, H_DIM);
        int tot = T_TOK * DINT;
        glu_k<<<(tot + 255) / 256, 256, 0, stream>>>(gubuf, actbf, tot);
        gemm_bf16_mxfp4<<<dim3((H_DIM + BN - 1) / BN, T_TOK / BM), 256, 0, stream>>>(
            actbf, dn_blocks + e * dnBlkStride, dn_scales + e * dnSclStride,
            dn_bias + (size_t)e * H_DIM, rsbuf, e,
            out, T_TOK, H_DIM, DINT);
    }
}
